// STFFilterLayer_64467459113195
// MI455X (gfx1250) — matrix operand order repaired and checked by reading, not compiled, not run on hardware
//
#include <hip/hip_runtime.h>
#include <hip/hip_bf16.h>

typedef _Float16 half_t;
typedef __attribute__((ext_vector_type(16))) _Float16 v16h;
typedef __attribute__((ext_vector_type(8)))  _Float16 v8h;
typedef __attribute__((ext_vector_type(4)))  _Float16 v4h;
typedef __attribute__((ext_vector_type(8)))  float    v8f;

#define ALPHA 0.2f
#define B_SZ 8
#define T_SZ 8192
#define C_OUT_TOT 96

// ---------------- WMMA helpers (CDNA5 layouts, wave32) ----------------

__device__ __forceinline__ v8f wmma_f16(v16h a, v16h b, v8f c) {
  // D = A(16x32 f16) x B(32x16 f16) + C(16x16 f32)
  return __builtin_amdgcn_wmma_f32_16x16x32_f16(
      false, a, false, b, (short)0, c, false, false);
}

// A fragment: row-major matrix [rows x lda], tile origin (row0, k0).
// Lane L: m = L&15, h = L>>4. Element e: K = (e&7) + 8*(2*(e>>3) + h).
// => two contiguous 8-half (16B) chunks at k0+8h and k0+16+8h.
__device__ __forceinline__ v16h load_a16(const half_t* A, int lda, int row0,
                                         int k0, int lane) {
  int m = lane & 15, h = lane >> 4;
  const half_t* base = A + (size_t)(row0 + m) * lda + k0 + 8 * h;
  v8h lo = *(const v8h*)(base);
  v8h hi = *(const v8h*)(base + 16);
  v16h r;
#pragma unroll
  for (int e = 0; e < 8; ++e) { r[e] = lo[e]; r[8 + e] = hi[e]; }
  return r;
}

// B fragment from COLUMN-major Bt [cols x ldb] (contraction contiguous).
// Lane L: n = L&15, h = L>>4. Element e: K = k0 + (e&7) + 8*(2*(e>>3) + h).
__device__ __forceinline__ v16h load_b16(const half_t* Bt, int ldb, int col0,
                                         int k0, int lane) {
  int n = lane & 15, h = lane >> 4;
  const half_t* base = Bt + (size_t)(col0 + n) * ldb + k0 + 8 * h;
  v8h lo = *(const v8h*)(base);
  v8h hi = *(const v8h*)(base + 16);
  v16h r;
#pragma unroll
  for (int e = 0; e < 8; ++e) { r[e] = lo[e]; r[8 + e] = hi[e]; }
  return r;
}

__device__ __forceinline__ float leaky(float v) {
  return v > 0.f ? v : ALPHA * v;
}

// ---------------- workspace layout (units: halfs) ----------------
// Branches: 0:(N=32,F=256,bins=17,Kp=32) 1:(64,128,33,64) 2:(128,64,65,96)
// W  [Kp x N] (Wr then Wi)         : b0@0      b1@2048    b2@10240   end 34816
// IW [N x 2Kp]                     : b0@34816  b1@36864   b2@45056   end 69632
// G  [Kp x 32 x 128] (Gtr then Gti): b0@69632  b1@331776  b2@856064  end 1642496
// X  [Kp x M x 128] f16 (reused)   : @1642496  (max 8388608 halfs)
// Y  [M x 32 x 2Kp] f16 (reused)   : @10031104 (max 4194304 halfs)
#define WS_W0 0
#define WS_W1 2048
#define WS_W2 10240
#define WS_IW0 34816
#define WS_IW1 36864
#define WS_IW2 45056
#define WS_G0 69632
#define WS_G1 331776
#define WS_G2 856064
#define WS_X 1642496
#define WS_Y 10031104

// ---------------- table construction ----------------

__global__ void init_tables_kernel(const float* __restrict__ kr,
                                   const float* __restrict__ ki,
                                   half_t* __restrict__ ws) {
  const int Nf[3] = {32, 64, 128};
  const int Kp[3] = {32, 64, 96};
  const int Bn[3] = {17, 33, 65};
  const size_t Woff[3] = {WS_W0, WS_W1, WS_W2};
  const size_t IWoff[3] = {WS_IW0, WS_IW1, WS_IW2};
  const size_t Goff[3] = {WS_G0, WS_G1, WS_G2};
  const float PI2 = 6.283185307179586f;

  int tid = blockIdx.x * blockDim.x + threadIdx.x;
  int nth = gridDim.x * blockDim.x;

  for (int br = 0; br < 3; ++br) {
    const int N = Nf[br], K = Kp[br], bins = Bn[br];
    // forward DFT: Wr[k][t]=cos(2pi kt/N), Wi[k][t]=-sin(...); zero-pad k>=bins
    for (int i = tid; i < K * N; i += nth) {
      int k = i / N, t = i % N;
      int kt = (k * t) % N;
      float th = PI2 * (float)kt / (float)N;
      float wr = (k < bins) ? __cosf(th) : 0.f;
      float wi = (k < bins) ? -__sinf(th) : 0.f;
      ws[Woff[br] + i] = (half_t)wr;
      ws[Woff[br] + (size_t)K * N + i] = (half_t)wi;
    }
    // inverse DFT combined [N x 2Kp]: cols [0,Kp)=Re weight, [Kp,2Kp)=Im weight
    for (int i = tid; i < N * 2 * K; i += nth) {
      int t = i / (2 * K), j = i % (2 * K);
      int k = (j < K) ? j : j - K;
      float v = 0.f;
      if (k < bins) {
        float w = (k == 0 || k == N / 2) ? 1.f : 2.f;
        int kt = (k * t) % N;
        float th = PI2 * (float)kt / (float)N;
        v = (j < K) ? (w * __cosf(th) / (float)N) : (-w * __sinf(th) / (float)N);
      }
      ws[IWoff[br] + i] = (half_t)v;
    }
    // filter: bilinear half-pixel resize 33 -> bins, combined complex form:
    // Gtr[k][o][j] = j<64 ?  Gr(k,j,o)   : -Gi(k,j-64,o)   (for Yr)
    // Gti[k][o][j] = j<64 ?  Gi(k,j,o)   :  Gr(k,j-64,o)   (for Yi)
    float scale = 33.f / (float)bins;
    for (int i = tid; i < K * 32 * 128; i += nth) {
      int k = i / 4096, rem = i % 4096, o = rem / 128, j = rem % 128;
      float gr = 0.f, gi = 0.f;
      if (k < bins) {
        int c = (j < 64) ? j : j - 64;
        float s = ((float)k + 0.5f) * scale - 0.5f;
        s = fminf(fmaxf(s, 0.f), 32.f);
        int i0 = (int)floorf(s);
        if (i0 > 32) i0 = 32;
        int i1 = (i0 + 1 > 32) ? 32 : i0 + 1;
        float fr = s - (float)i0;
        int idx0 = i0 * 2048 + c * 32 + o;
        int idx1 = i1 * 2048 + c * 32 + o;
        gr = kr[idx0] * (1.f - fr) + kr[idx1] * fr;
        gi = ki[idx0] * (1.f - fr) + ki[idx1] * fr;
      }
      bool lo = (j < 64);
      ws[Goff[br] + i] = (half_t)(lo ? gr : -gi);
      ws[Goff[br] + (size_t)K * 4096 + i] = (half_t)(lo ? gi : gr);
    }
  }
}

// ---------------- stage A: per-frame rDFT via WMMA (transposed D) ----------
// grid = B*FR blocks, 32 threads. Computes X^T[c][k] = xs[c][t] * W^T[t][k].
// Output X[k][bf][0:64]=Re, [64:128]=Im (f16); lane rows (c) are stride-1 in
// X so each accumulator stores as one 16-byte v8h.
template <int NFFT, int KP, int FR>
__global__ void stage_a_kernel(const float* __restrict__ x,
                               const half_t* __restrict__ W,
                               half_t* __restrict__ X) {
  __shared__ __align__(16) half_t xs[64 * NFFT];  // [c][t]
  const int bf = blockIdx.x;
  const int b = bf / FR, f = bf % FR;
  const int lane = threadIdx.x & 31;
  const int M = B_SZ * FR;

  // transpose fill: thread handles (c, 4 consecutive t) -> one ds_store_b64
  const float* xp = x + ((size_t)b * T_SZ + (size_t)f * NFFT) * 64;
  for (int i = threadIdx.x; i < (NFFT / 4) * 64; i += 32) {
    int c = i & 63, t4 = i >> 6;
    v4h p;
#pragma unroll
    for (int u = 0; u < 4; ++u)
      p[u] = (half_t)xp[(size_t)(t4 * 4 + u) * 64 + c];
    *(v4h*)(&xs[c * NFFT + t4 * 4]) = p;
  }
  __syncthreads();

  const half_t* Wr = W;                       // [KP][NFFT] row-major
  const half_t* Wi = W + (size_t)KP * NFFT;   // == col-major B over t
  const int n = lane & 15, h = lane >> 4;

#pragma unroll
  for (int kt = 0; kt < KP / 16; ++kt) {
#pragma unroll
    for (int ct = 0; ct < 4; ++ct) {
      v8f accr = {}; v8f acci = {};
#pragma unroll
      for (int s = 0; s < NFFT / 32; ++s) {
        v16h a = load_a16(xs, NFFT, ct * 16, s * 32, lane);   // rows = c (LDS)
        v16h br = load_b16(Wr, NFFT, kt * 16, s * 32, lane);  // cols = k
        v16h bi = load_b16(Wi, NFFT, kt * 16, s * 32, lane);
        accr = wmma_f16(a, br, accr);
        acci = wmma_f16(a, bi, acci);
      }
      int k = kt * 16 + n;          // fixed per lane
      int c0 = ct * 16 + 8 * h;     // 8 consecutive c across accum regs
      size_t base = ((size_t)k * M + bf) * 128;
      v8h pr, pi;
#pragma unroll
      for (int r = 0; r < 8; ++r) {
        pr[r] = (half_t)accr[r];
        pi[r] = (half_t)acci[r];
      }
      *(v8h*)(X + base + c0) = pr;
      *(v8h*)(X + base + 64 + c0) = pi;
    }
  }
}

// ---------------- stage B: per-bin complex filter + leaky ----------------
// grid = (KP, M/16), 32 threads. Y[m][o][0:Kp]=Re, [Kp:2Kp]=Im (f16).
template <int KP, int M>
__global__ void stage_b_kernel(const half_t* __restrict__ X,
                               const half_t* __restrict__ Gtr,
                               const half_t* __restrict__ Gti,
                               half_t* __restrict__ Y) {
  const int k = blockIdx.x;
  const int mt = blockIdx.y;
  const int lane = threadIdx.x & 31;
  const half_t* Xk = X + (size_t)k * M * 128;
  const half_t* Gr = Gtr + (size_t)k * 32 * 128;
  const half_t* Gi = Gti + (size_t)k * 32 * 128;

  v8f yr0 = {}, yr1 = {}, yi0 = {}, yi1 = {};
#pragma unroll
  for (int s = 0; s < 4; ++s) {
    v16h af = load_a16(Xk, 128, mt * 16, s * 32, lane);
    v16h br0 = load_b16(Gr, 128, 0, s * 32, lane);
    v16h br1 = load_b16(Gr, 128, 16, s * 32, lane);
    v16h bi0 = load_b16(Gi, 128, 0, s * 32, lane);
    v16h bi1 = load_b16(Gi, 128, 16, s * 32, lane);
    yr0 = wmma_f16(af, br0, yr0);
    yr1 = wmma_f16(af, br1, yr1);
    yi0 = wmma_f16(af, bi0, yi0);
    yi1 = wmma_f16(af, bi1, yi1);
  }
  const int n = lane & 15, h = lane >> 4;
#pragma unroll
  for (int r = 0; r < 8; ++r) {
    int m = mt * 16 + r + 8 * h;
    size_t b0 = ((size_t)m * 32 + n) * (2 * KP);
    size_t b1 = ((size_t)m * 32 + 16 + n) * (2 * KP);
    Y[b0 + k] = (half_t)leaky(yr0[r]);
    Y[b1 + k] = (half_t)leaky(yr1[r]);
    Y[b0 + KP + k] = (half_t)leaky(yi0[r]);
    Y[b1 + KP + k] = (half_t)leaky(yi1[r]);
  }
}

// ---------------- stage C: irDFT + bias (transposed D) ----------------
// grid = B*FR blocks, 32 threads. Computes out^T[o][t] = Y[o][j] * IW^T[j][t];
// lane rows (o) are stride-1 in the output -> contiguous 32B f32 stores,
// bias becomes a contiguous vector load.
template <int NFFT, int KP, int FR>
__global__ void stage_c_kernel(const half_t* __restrict__ IW,
                               const half_t* __restrict__ Y,
                               const float* __restrict__ bias,
                               float* __restrict__ out, int obase) {
  const int bf = blockIdx.x;
  const int b = bf / FR, f = bf % FR;
  const int lane = threadIdx.x & 31;
  const int n = lane & 15, h = lane >> 4;
  const half_t* Yb = Y + (size_t)bf * 32 * (2 * KP);  // [32 o][2Kp] row-major
  float* op = out + ((size_t)b * T_SZ + (size_t)f * NFFT) * C_OUT_TOT + obase;

#pragma unroll
  for (int tt = 0; tt < NFFT / 16; ++tt) {
#pragma unroll
    for (int ot = 0; ot < 2; ++ot) {
      v8f acc = {};
#pragma unroll
      for (int s = 0; s < 2 * KP / 32; ++s) {
        v16h a = load_a16(Yb, 2 * KP, ot * 16, s * 32, lane);  // rows = o
        v16h bfrag = load_b16(IW, 2 * KP, tt * 16, s * 32, lane);  // cols = t
        acc = wmma_f16(a, bfrag, acc);
      }
      int t = tt * 16 + n;                         // fixed per lane
      const float* bp = bias + obase + ot * 16 + 8 * h;
      float* p = op + (size_t)t * C_OUT_TOT + ot * 16 + 8 * h;
#pragma unroll
      for (int r = 0; r < 8; ++r) p[r] = acc[r] + bp[r];
    }
  }
}

// ---------------- launch ----------------

extern "C" void kernel_launch(void* const* d_in, const int* in_sizes, int n_in,
                              void* d_out, int out_size, void* d_ws,
                              size_t ws_size, hipStream_t stream) {
  const float* x = (const float*)d_in[0];        // [8,8192,64]
  const float* kernel_r = (const float*)d_in[1]; // [1,1,33,2048]
  const float* kernel_i = (const float*)d_in[2]; // [1,1,33,2048]
  const float* bias_r = (const float*)d_in[3];   // [96]
  float* out = (float*)d_out;                    // [8,8192,96]
  half_t* ws = (half_t*)d_ws;                    // needs ~28.5 MB

  init_tables_kernel<<<512, 256, 0, stream>>>(kernel_r, kernel_i, ws);

  half_t* X = ws + WS_X;
  half_t* Y = ws + WS_Y;

  // ---- branch 0: N=32, F=256, Kp=32, M=2048, obase=0 ----
  stage_a_kernel<32, 32, 256><<<2048, 32, 0, stream>>>(x, ws + WS_W0, X);
  stage_b_kernel<32, 2048><<<dim3(32, 128), 32, 0, stream>>>(
      X, ws + WS_G0, ws + WS_G0 + (size_t)32 * 4096, Y);
  stage_c_kernel<32, 32, 256><<<2048, 32, 0, stream>>>(ws + WS_IW0, Y, bias_r,
                                                       out, 0);

  // ---- branch 1: N=64, F=128, Kp=64, M=1024, obase=32 ----
  stage_a_kernel<64, 64, 128><<<1024, 32, 0, stream>>>(x, ws + WS_W1, X);
  stage_b_kernel<64, 1024><<<dim3(64, 64), 32, 0, stream>>>(
      X, ws + WS_G1, ws + WS_G1 + (size_t)64 * 4096, Y);
  stage_c_kernel<64, 64, 128><<<1024, 32, 0, stream>>>(ws + WS_IW1, Y, bias_r,
                                                       out, 32);

  // ---- branch 2: N=128, F=64, Kp=96, M=512, obase=64 ----
  stage_a_kernel<128, 96, 64><<<512, 32, 0, stream>>>(x, ws + WS_W2, X);
  stage_b_kernel<96, 512><<<dim3(96, 32), 32, 0, stream>>>(
      X, ws + WS_G2, ws + WS_G2 + (size_t)96 * 4096, Y);
  stage_c_kernel<128, 96, 64><<<512, 32, 0, stream>>>(ws + WS_IW2, Y, bias_r,
                                                      out, 64);
}
